// HierarchicalGraphNeuralNetwork_11982958756658
// MI455X (gfx1250) — matrix operand order repaired and checked by reading, not compiled, not run on hardware
//
#include <hip/hip_runtime.h>
#include <hip/hip_bf16.h>
#include <math.h>

// ---------------- problem constants (match reference) ----------------
#define N_CFG   400000
#define E_CFG   1600000
#define N_FUNC  8000
#define N_EXT   1600
#define N_FCGN  (N_FUNC + N_EXT)   // 9600
#define E_FCG   80000
#define NBATCH  8
#define D_IN    64
#define D_H     128
#define VOCAB   10002

typedef __attribute__((ext_vector_type(16))) __bf16 v16bf;
typedef __attribute__((ext_vector_type(8)))  float  v8f;

// ---------------- WMMA fragment packing helpers (ISA 7.12.2, wave32) ----------------
// A-fragment: lane (m, half) of tile `tile`, K-chunk kt holds 16 contiguous bf16:
//   j in [0,8)  -> K = 32*kt +      8*half + j
//   j in [8,16) -> K = 32*kt + 16 + 8*half + (j-8)
// Packed A element offset for logical (node, channel c), KT = K/32 chunks:
__device__ __forceinline__ long a_pack_off(int node, int c, int KT) {
  int tile = node >> 4, m = node & 15;
  int kt   = c >> 5,    r = c & 31;
  int half = (r >> 3) & 1;
  int j    = ((r >> 4) << 3) | (r & 7);
  int lane = (half << 4) | m;
  return ((((long)tile * KT + kt) * 32 + lane) << 4) + j;
}

// ---------------- utility kernels ----------------
__global__ void k_zero(float* __restrict__ p, long n) {
  long i = (long)blockIdx.x * blockDim.x + threadIdx.x;
  long stride = (long)gridDim.x * blockDim.x;
  for (; i < n; i += stride) p[i] = 0.0f;
}

__global__ void k_count_deg(const int* __restrict__ col, float* __restrict__ deg, int e) {
  int i = blockIdx.x * blockDim.x + threadIdx.x;
  if (i < e) atomicAdd(&deg[col[i]], 1.0f);
}

__global__ void k_rsqrt_deg(float* __restrict__ d, int n) {
  int i = blockIdx.x * blockDim.x + threadIdx.x;
  if (i < n) d[i] = rsqrtf(d[i] + 1.0f);   // deg + self-loop
}

// ---------------- pack fp32 matrix [M,K] -> A-fragment bf16 ----------------
__global__ void k_pack_a(const float* __restrict__ X, __bf16* __restrict__ Ap,
                         long total, int Kshift) {
  long i = (long)blockIdx.x * blockDim.x + threadIdx.x;
  long stride = (long)gridDim.x * blockDim.x;
  int K = 1 << Kshift, KT = K >> 5;
  for (; i < total; i += stride) {
    int node = (int)(i >> Kshift);
    int c    = (int)(i & (K - 1));
    Ap[a_pack_off(node, c, KT)] = (__bf16)X[i];
  }
}

// ---------------- pack fp32 W [K,128] -> B-fragment bf16 ----------------
// B-fragment (kt, nt): lane (n, half) holds 16 contiguous bf16:
//   j in [0,8)  -> W[32*kt +      8*half + j][16*nt + n]
//   j in [8,16) -> W[32*kt + 16 + 8*half + (j-8)][16*nt + n]
__global__ void k_pack_w(const float* __restrict__ W, __bf16* __restrict__ Bp, int K) {
  int i = blockIdx.x * blockDim.x + threadIdx.x;
  if (i >= K * D_H) return;
  int k = i >> 7, col = i & 127;
  int kt = k >> 5, r = k & 31;
  int half = (r >> 3) & 1, j = ((r >> 4) << 3) | (r & 7);
  int nt = col >> 4, n = col & 15;
  int lane = (half << 4) | n;
  long off = ((((long)kt * 8 + nt) * 32 + lane) << 4) + j;
  Bp[off] = (__bf16)W[i];
}

// ---------------- WMMA GEMM: H[M,128] = Apack[M,K] @ Bpack[K,128] ----------------
// One wave per 16x128 tile; per kt: 2 b128 A-loads, 16 b128 B-loads, 8 WMMA.
__global__ void k_gemm_wmma(const __bf16* __restrict__ Ap, const __bf16* __restrict__ Bp,
                            float* __restrict__ H, int M, int KT) {
  int lane = threadIdx.x & 31;
  int tile = blockIdx.x * (blockDim.x >> 5) + (threadIdx.x >> 5);
  int row0 = tile << 4;
  if (row0 >= M) return;                   // uniform per wave; EXEC stays full
  int m = lane & 15, half = lane >> 4;
  const v16bf* A = (const v16bf*)Ap + (long)tile * KT * 32 + lane;
  const v16bf* B = (const v16bf*)Bp + lane;

  v8f acc[8] = {};
  for (int kt = 0; kt < KT; ++kt) {
    v16bf a = A[(long)kt * 32];
#pragma unroll
    for (int nt = 0; nt < 8; ++nt) {
      v16bf b = B[(long)(kt * 8 + nt) * 32];
      acc[nt] = __builtin_amdgcn_wmma_f32_16x16x32_bf16(
          false, a, false, b, (short)0, acc[nt], false, false);
    }
  }
  // C/D layout: VGPR i -> row (i + 8*half), col = m within n-tile
#pragma unroll
  for (int nt = 0; nt < 8; ++nt)
#pragma unroll
    for (int i = 0; i < 8; ++i)
      H[(long)(row0 + i + (half << 3)) * D_H + (nt << 4) + m] = acc[nt][i];
}

// ---------------- edge scatter: agg[col] += h[row] * dis[row]*dis[col] ----------------
__global__ void k_edge_scatter(const float* __restrict__ H, const int* __restrict__ rowi,
                               const int* __restrict__ coli, const float* __restrict__ dis,
                               float* __restrict__ agg, int e) {
  int lane = threadIdx.x & 31;
  int eid  = blockIdx.x * (blockDim.x >> 5) + (threadIdx.x >> 5);
  if (eid >= e) return;
  int r = rowi[eid], c = coli[eid];
  float nrm = dis[r] * dis[c];
  const float4* hs = (const float4*)(H + (long)r * D_H);
  float4 v = hs[lane];
  float* dst = agg + (long)c * D_H + lane * 4;
  atomicAdd(dst + 0, v.x * nrm);
  atomicAdd(dst + 1, v.y * nrm);
  atomicAdd(dst + 2, v.z * nrm);
  atomicAdd(dst + 3, v.w * nrm);
}

// ---------------- relu(agg + h*dis^2 + bias) -> fp32 (in-place capable) ----------------
__global__ void k_gcn_finalize(const float* __restrict__ agg, const float* __restrict__ H,
                               const float* __restrict__ dis, const float* __restrict__ bias,
                               float* __restrict__ out, long n_nodes) {
  long total = n_nodes * D_H;
  long stride = (long)gridDim.x * blockDim.x;
  for (long i = (long)blockIdx.x * blockDim.x + threadIdx.x; i < total; i += stride) {
    long node = i >> 7;
    int  c    = (int)(i & 127);
    float d = dis[node];
    float v = agg[i] + H[i] * (d * d) + bias[c];
    out[i] = v > 0.0f ? v : 0.0f;
  }
}

// ---------------- relu(agg + h*dis^2 + bias) -> packed bf16 A (next GEMM, K=128) ----------------
__global__ void k_gcn_finalize_pack(const float* __restrict__ agg, const float* __restrict__ H,
                                    const float* __restrict__ dis, const float* __restrict__ bias,
                                    __bf16* __restrict__ Ap, long n_nodes) {
  long total = n_nodes * D_H;
  long stride = (long)gridDim.x * blockDim.x;
  for (long i = (long)blockIdx.x * blockDim.x + threadIdx.x; i < total; i += stride) {
    int node = (int)(i >> 7);
    int c    = (int)(i & 127);
    float d = dis[node];
    float v = agg[i] + H[i] * (d * d) + bias[c];
    Ap[a_pack_off(node, c, 4)] = (__bf16)(v > 0.0f ? v : 0.0f);
  }
}

// ---------------- segment sum (warp per node) + divide ----------------
__global__ void k_seg_sum(const float* __restrict__ X, const int* __restrict__ seg,
                          float* __restrict__ sum, float* __restrict__ cnt, int n) {
  int lane = threadIdx.x & 31;
  int node = blockIdx.x * (blockDim.x >> 5) + (threadIdx.x >> 5);
  if (node >= n) return;
  int s = seg[node];
  const float4* xs = (const float4*)(X + (long)node * D_H);
  float4 v = xs[lane];
  float* dst = sum + (long)s * D_H + lane * 4;
  atomicAdd(dst + 0, v.x);
  atomicAdd(dst + 1, v.y);
  atomicAdd(dst + 2, v.z);
  atomicAdd(dst + 3, v.w);
  if (lane == 0) atomicAdd(&cnt[s], 1.0f);
}

__global__ void k_seg_div(float* __restrict__ sum, const float* __restrict__ cnt, int n) {
  int i = blockIdx.x * blockDim.x + threadIdx.x;
  if (i < n * D_H) {
    int s = i >> 7;
    sum[i] /= fmaxf(cnt[s], 1.0f);
  }
}

// ---------------- assemble FCG node features -> packed bf16 A (gemm3, K=128) ----------------
__global__ void k_assemble_pack(const float* __restrict__ pooled, const float* __restrict__ emb,
                                const int* __restrict__ src, const int* __restrict__ isext,
                                __bf16* __restrict__ Ap, int n) {
  int i = blockIdx.x * blockDim.x + threadIdx.x;
  if (i >= n * D_H) return;
  int node = i >> 7, c = i & 127;
  int s = src[node];
  float v;
  if (isext[node] == 1) {
    int idx = min(max(s, 0), VOCAB - 1);           // mode='clip'
    v = emb[(long)idx * D_H + c];
  } else {
    int idx = min(max(s, 0), N_FUNC - 1);          // jnp.clip
    v = pooled[(long)idx * D_H + c];
  }
  Ap[a_pack_off(node, c, 4)] = (__bf16)v;
}

// ---------------- projection head, single block ----------------
__global__ void k_head(const float* __restrict__ g,
                       const float* __restrict__ Wp1, const float* __restrict__ bp1,
                       const float* __restrict__ Wp2, const float* __restrict__ bp2,
                       const float* __restrict__ Wp3, const float* __restrict__ bp3,
                       float* __restrict__ out) {
  __shared__ float sg[NBATCH * 128];
  __shared__ float sh1[NBATCH * 64];
  __shared__ float sh2[NBATCH * 32];
  int tid = threadIdx.x;
  for (int i = tid; i < NBATCH * 128; i += blockDim.x) sg[i] = g[i];
  __syncthreads();
  for (int o = tid; o < NBATCH * 64; o += blockDim.x) {
    int b = o >> 6, j = o & 63;
    float s = bp1[j];
    for (int k = 0; k < 128; ++k) s += sg[b * 128 + k] * Wp1[k * 64 + j];
    sh1[o] = fmaxf(s, 0.0f);
  }
  __syncthreads();
  for (int o = tid; o < NBATCH * 32; o += blockDim.x) {
    int b = o >> 5, j = o & 31;
    float s = bp2[j];
    for (int k = 0; k < 64; ++k) s += sh1[b * 64 + k] * Wp2[k * 32 + j];
    sh2[o] = fmaxf(s, 0.0f);
  }
  __syncthreads();
  if (tid < NBATCH) {
    float s = bp3[0];
    for (int k = 0; k < 32; ++k) s += sh2[tid * 32 + k] * Wp3[k];
    out[tid] = 1.0f / (1.0f + expf(-s));
  }
}

// ---------------- host orchestration ----------------
extern "C" void kernel_launch(void* const* d_in, const int* in_sizes, int n_in,
                              void* d_out, int out_size, void* d_ws, size_t ws_size,
                              hipStream_t stream) {
  (void)in_sizes; (void)n_in; (void)out_size; (void)ws_size;
  const float* cfg_x      = (const float*)d_in[0];
  const int*   cfg_ei     = (const int*)  d_in[1];   // [2, E_CFG]
  const int*   node2func  = (const int*)  d_in[2];
  const int*   fcg_ei     = (const int*)  d_in[3];   // [2, E_FCG]
  const int*   fcg_batch  = (const int*)  d_in[4];
  const int*   fcg_src    = (const int*)  d_in[5];
  const int*   fcg_isext  = (const int*)  d_in[6];
  const float* W_cfg1     = (const float*)d_in[7];
  const float* b_cfg1     = (const float*)d_in[8];
  const float* W_cfg2     = (const float*)d_in[9];
  const float* b_cfg2     = (const float*)d_in[10];
  const float* emb_table  = (const float*)d_in[11];
  const float* W_fcg1     = (const float*)d_in[12];
  const float* b_fcg1     = (const float*)d_in[13];
  const float* Wp1        = (const float*)d_in[14];
  const float* bp1        = (const float*)d_in[15];
  const float* Wp2        = (const float*)d_in[16];
  const float* bp2        = (const float*)d_in[17];
  const float* Wp3        = (const float*)d_in[18];
  const float* bp3        = (const float*)d_in[19];
  float* out = (float*)d_out;

  // workspace layout (float units; every block a multiple of 16 floats -> 32B aligned)
  float* w = (float*)d_ws;
  size_t o = 0;
  float*  dis_cfg = w + o; o += N_CFG;
  float*  hbuf    = w + o; o += (size_t)N_CFG * D_H;
  float*  aggbuf  = w + o; o += (size_t)N_CFG * D_H;
  float*  pooled  = w + o; o += (size_t)N_FUNC * D_H;
  float*  cntf    = w + o; o += N_FUNC;
  float*  dis_fcg = w + o; o += N_FCGN;
  float*  fcgh    = w + o; o += (size_t)N_FCGN * D_H;
  float*  fcgagg  = w + o; o += (size_t)N_FCGN * D_H;
  float*  gbuf    = w + o; o += NBATCH * D_H;
  float*  cntb    = w + o; o += 16;
  __bf16* apack1  = (__bf16*)(w + o); o += (size_t)N_CFG * D_IN / 2;   // bf16 = half a float
  __bf16* apack2  = (__bf16*)(w + o); o += (size_t)N_CFG * D_H / 2;
  __bf16* apack3  = (__bf16*)(w + o); o += (size_t)N_FCGN * D_H / 2;
  __bf16* bpack1  = (__bf16*)(w + o); o += (D_IN * D_H) / 2;
  __bf16* bpack2  = (__bf16*)(w + o); o += (D_H * D_H) / 2;
  __bf16* bpack3  = (__bf16*)(w + o); o += (D_H * D_H) / 2;

  const int T = 256;

  // --- pack weights (tiny, L2-resident) ---
  k_pack_w<<<(D_IN * D_H + T - 1) / T, T, 0, stream>>>(W_cfg1, bpack1, D_IN);
  k_pack_w<<<(D_H * D_H + T - 1) / T, T, 0, stream>>>(W_cfg2, bpack2, D_H);
  k_pack_w<<<(D_H * D_H + T - 1) / T, T, 0, stream>>>(W_fcg1, bpack3, D_H);

  // --- pack cfg_x into A fragments (fp32 -> bf16, halves gemm1 A traffic) ---
  k_pack_a<<<8192, T, 0, stream>>>(cfg_x, apack1, (long)N_CFG * D_IN, 6);

  // --- CFG degrees ---
  k_zero<<<1024, T, 0, stream>>>(dis_cfg, (long)N_CFG);
  k_count_deg<<<(E_CFG + T - 1) / T, T, 0, stream>>>(cfg_ei + E_CFG, dis_cfg, E_CFG);
  k_rsqrt_deg<<<(N_CFG + T - 1) / T, T, 0, stream>>>(dis_cfg, N_CFG);

  // --- CFG GCN layer 1 ---
  k_gemm_wmma<<<(N_CFG / 16) / 8, T, 0, stream>>>(apack1, bpack1, hbuf, N_CFG, D_IN / 32);
  k_zero<<<4096, T, 0, stream>>>(aggbuf, (long)N_CFG * D_H);
  k_edge_scatter<<<(E_CFG + 7) / 8, T, 0, stream>>>(hbuf, cfg_ei, cfg_ei + E_CFG, dis_cfg, aggbuf, E_CFG);
  // x1 lives only as gemm2's A operand -> emit packed bf16 directly
  k_gcn_finalize_pack<<<4096, T, 0, stream>>>(aggbuf, hbuf, dis_cfg, b_cfg1, apack2, (long)N_CFG);

  // --- CFG GCN layer 2 ---
  k_gemm_wmma<<<(N_CFG / 16) / 8, T, 0, stream>>>(apack2, bpack2, hbuf, N_CFG, D_H / 32);
  k_zero<<<4096, T, 0, stream>>>(aggbuf, (long)N_CFG * D_H);
  k_edge_scatter<<<(E_CFG + 7) / 8, T, 0, stream>>>(hbuf, cfg_ei, cfg_ei + E_CFG, dis_cfg, aggbuf, E_CFG);
  k_gcn_finalize<<<4096, T, 0, stream>>>(aggbuf, hbuf, dis_cfg, b_cfg2, aggbuf, (long)N_CFG);

  // --- per-function mean pool ---
  k_zero<<<1024, T, 0, stream>>>(pooled, (long)N_FUNC * D_H);
  k_zero<<<32, T, 0, stream>>>(cntf, (long)N_FUNC);
  k_seg_sum<<<(N_CFG + 7) / 8, T, 0, stream>>>(aggbuf, node2func, pooled, cntf, N_CFG);
  k_seg_div<<<(N_FUNC * D_H + T - 1) / T, T, 0, stream>>>(pooled, cntf, N_FUNC);

  // --- assemble FCG features straight into gemm3's packed A ---
  k_assemble_pack<<<(N_FCGN * D_H + T - 1) / T, T, 0, stream>>>(pooled, emb_table, fcg_src, fcg_isext, apack3, N_FCGN);

  // --- FCG degrees + GCN ---
  k_zero<<<64, T, 0, stream>>>(dis_fcg, (long)N_FCGN);
  k_count_deg<<<(E_FCG + T - 1) / T, T, 0, stream>>>(fcg_ei + E_FCG, dis_fcg, E_FCG);
  k_rsqrt_deg<<<(N_FCGN + T - 1) / T, T, 0, stream>>>(dis_fcg, N_FCGN);
  k_gemm_wmma<<<(N_FCGN / 16) / 8, T, 0, stream>>>(apack3, bpack3, fcgh, N_FCGN, D_H / 32);
  k_zero<<<1024, T, 0, stream>>>(fcgagg, (long)N_FCGN * D_H);
  k_edge_scatter<<<(E_FCG + 7) / 8, T, 0, stream>>>(fcgh, fcg_ei, fcg_ei + E_FCG, dis_fcg, fcgagg, E_FCG);
  k_gcn_finalize<<<512, T, 0, stream>>>(fcgagg, fcgh, dis_fcg, b_fcg1, fcgagg, (long)N_FCGN);

  // --- per-binary mean pool ---
  k_zero<<<8, T, 0, stream>>>(gbuf, (long)(NBATCH * D_H));
  k_zero<<<1, 32, 0, stream>>>(cntb, (long)NBATCH);
  k_seg_sum<<<(N_FCGN + 7) / 8, T, 0, stream>>>(fcgagg, fcg_batch, gbuf, cntb, N_FCGN);
  k_seg_div<<<(NBATCH * D_H + T - 1) / T, T, 0, stream>>>(gbuf, cntb, NBATCH);

  // --- projection head + sigmoid ---
  k_head<<<1, 128, 0, stream>>>(gbuf, Wp1, bp1, Wp2, bp2, Wp3, bp3, out);
}
